// ODESurvMultiple_7602092114332
// MI455X (gfx1250) — matrix operand order repaired and checked by reading, not compiled, not run on hardware
//
#include <hip/hip_runtime.h>

// ---------------------------------------------------------------------------
// ODESurvMultiple fused forward for gfx1250 (MI455X), bf16 WMMA everywhere.
// B=16384, COV=256, H=512, K=8, NQ=15.
// ---------------------------------------------------------------------------

typedef float v8f __attribute__((ext_vector_type(8)));
typedef float v4f __attribute__((ext_vector_type(4)));
typedef int v4i __attribute__((ext_vector_type(4)));
typedef unsigned short v16u __attribute__((ext_vector_type(16)));
typedef __bf16 v16bf __attribute__((ext_vector_type(16)));

#define WMMA_BF16(a, b, c) \
  __builtin_amdgcn_wmma_f32_16x16x32_bf16(false, (a), false, (b), (short)0, (c), false, false)

// ws layout (bytes)
#define WS_W1P 0u           // 8 kchunks * 32 ntiles * 1024B
#define WS_W1O 262144u
#define WS_W2P 524288u      // 16 kchunks * 1024B (N padded 8->16)
#define WS_W2O 540672u
#define WS_WL  557056u      // 512 f32
// total 559104 bytes

__constant__ float UQ[15] = {
    -0.9879925180204854f, -0.9372733924007060f, -0.8482065834104272f,
    -0.7244177313601701f, -0.5709721726085388f, -0.3941513470775634f,
    -0.2011940939974345f,  0.0f,                 0.2011940939974345f,
     0.3941513470775634f,  0.5709721726085388f,  0.7244177313601701f,
     0.8482065834104272f,  0.9372733924007060f,  0.9879925180204854f};
__constant__ float WQ[15] = {
    0.0307532419961173f, 0.0703660474881081f, 0.1071592204671719f,
    0.1395706779261543f, 0.1662692058169939f, 0.1861610000155622f,
    0.1984314853271116f, 0.2025782419255613f, 0.1984314853271116f,
    0.1861610000155622f, 0.1662692058169939f, 0.1395706779261543f,
    0.1071592204671719f, 0.0703660474881081f, 0.0307532419961173f};

__device__ __forceinline__ unsigned short f2bf(float f) {
  unsigned int b = __float_as_uint(f);
  b += 0x7FFFu + ((b >> 16) & 1u);   // round-to-nearest-even
  return (unsigned short)(b >> 16);
}
__device__ __forceinline__ float bf2f(unsigned int u) {
  return __uint_as_float(u << 16);
}

struct Frag32B { v4i a, b; };   // 32 bytes == one v16bf

__device__ __forceinline__ v16bf load_frag(const unsigned char* p) {
  Frag32B pr;
  pr.a = *(const v4i*)p;
  pr.b = *(const v4i*)(p + 16);
  return __builtin_bit_cast(v16bf, pr);
}

// ---------------------------------------------------------------------------
// Prep: convert weights to bf16 pre-swizzled into WMMA B-fragment order.
// B-frag (32x16 bf16 tile): lane L, elem i  ->  W[k0 + (i<8 ? i : i+8) + 8*(L/16), n0 + (L%16)]
// ---------------------------------------------------------------------------
__global__ __launch_bounds__(256) void odesurv_prep(
    const float* __restrict__ W1p, const float* __restrict__ W2p,
    const float* __restrict__ W1o, const float* __restrict__ W2o,
    unsigned char* __restrict__ ws) {
  const int gid = blockIdx.x * 256 + threadIdx.x;
  const int E1 = 8 * 32 * 32 * 16;   // 131072 per big weight
  const int E3 = 16 * 32 * 16;       // 8192 per small weight
  if (gid < 2 * E1) {
    const int e = gid & (E1 - 1);
    const int which = gid >> 17;
    const int i = e & 15, L = (e >> 4) & 31, tl = e >> 9;
    const int nt = tl & 31, kc = tl >> 5;
    const int k = kc * 32 + (i < 8 ? i : i + 8) + 8 * (L >> 4);
    const int n = nt * 16 + (L & 15);
    const float* src = which ? W1o : W1p;
    unsigned short* dst = (unsigned short*)(ws + (which ? WS_W1O : WS_W1P));
    dst[e] = f2bf(src[k * 512 + n]);
  } else if (gid < 2 * E1 + 2 * E3) {
    const int e2 = gid - 2 * E1;
    const int e = e2 & (E3 - 1);
    const int which = e2 >> 13;
    const int i = e & 15, L = (e >> 4) & 31, kc = e >> 9;
    const int k = kc * 32 + (i < 8 ? i : i + 8) + 8 * (L >> 4);
    const int n = L & 15;
    const float* src = which ? W2o : W2p;
    unsigned short* dst = (unsigned short*)(ws + (which ? WS_W2O : WS_W2P));
    dst[e] = f2bf((n < 8) ? src[k * 8 + n] : 0.0f);
  } else if (gid < 2 * E1 + 2 * E3 + 512) {
    const int j = gid - (2 * E1 + 2 * E3);
    ((float*)(ws + WS_WL))[j] = W1o[256 * 512 + j];  // last row of W1o
  }
}

// ---------------------------------------------------------------------------
// Main: one wave == one 16-row tile. 2 waves / block.
// ---------------------------------------------------------------------------
__global__ __launch_bounds__(64) void odesurv_main(
    const float* __restrict__ x, const float* __restrict__ t,
    const float* __restrict__ b1p, const float* __restrict__ b2p,
    const float* __restrict__ b1o, const float* __restrict__ b2o,
    const unsigned char* __restrict__ ws, float* __restrict__ out) {
  constexpr int HXP = 520;                        // bf16 pitch: bank-offset 4/row
  __shared__ unsigned short hxs[2][16 * HXP];     // hx staging, bf16, per wave
  __shared__ unsigned short h1s[2][16 * 40];      // h1 restage (16x32, pitch 40)

  const int wave = threadIdx.x >> 5;
  const int lane = threadIdx.x & 31;
  const int tile = blockIdx.x * 2 + wave;
  const int row0 = tile * 16;
  const int mA = lane & 15;       // A-fragment row for this lane
  const int halfk = lane >> 4;    // 0/1: which K half of chunk
  const int o = halfk * 8;        // K offset of this lane's first run
  const int nC = lane & 15;       // C-layout column for this lane

  unsigned short* hxb = hxs[wave];
  unsigned short* h1b = h1s[wave];
  const unsigned char* fW1p = ws + WS_W1P;
  const unsigned char* fW1o = ws + WS_W1O;
  const unsigned char* fW2p = ws + WS_W2P;
  const unsigned char* fW2o = ws + WS_W2O;
  const float* wlast = (const float*)(ws + WS_WL);

  // ---- load x tile into 8 bf16 A-fragments (K chunks of 32) ----
  // A layout: lane L elem i -> k = kc*32 + (i<8 ? i : i+8) + 8*(L/16)
  v16bf xa[8];
  const float* xrow = x + (size_t)(row0 + mA) * 256;
#pragma unroll
  for (int kc = 0; kc < 8; ++kc) {
    const float* p = xrow + kc * 32 + o;
    v4f f0 = *(const v4f*)(p);
    v4f f1 = *(const v4f*)(p + 4);
    v4f f2 = *(const v4f*)(p + 16);
    v4f f3 = *(const v4f*)(p + 20);
    v16u u;
#pragma unroll
    for (int j = 0; j < 4; ++j) {
      u[j] = f2bf(f0[j]);
      u[4 + j] = f2bf(f1[j]);
      u[8 + j] = f2bf(f2[j]);
      u[12 + j] = f2bf(f3[j]);
    }
    xa[kc] = __builtin_bit_cast(v16bf, u);
  }

  // ---- stage 1: h1 = relu(x@W1p+b1p), hx = x@W1o+b1o, acc2p = h1@W2p ----
  v8f acc2p = {0, 0, 0, 0, 0, 0, 0, 0};
  for (int c = 0; c < 16; ++c) {
#pragma unroll
    for (int sub = 0; sub < 2; ++sub) {
      const int nt = 2 * c + sub;
      v8f a1 = {0, 0, 0, 0, 0, 0, 0, 0};
      v8f ao = {0, 0, 0, 0, 0, 0, 0, 0};
      const unsigned char* bp = fW1p + (size_t)nt * 1024 + lane * 32;
      const unsigned char* bo = fW1o + (size_t)nt * 1024 + lane * 32;
#pragma unroll
      for (int kc = 0; kc < 8; ++kc) {
        a1 = WMMA_BF16(xa[kc], load_frag(bp + (size_t)kc * 32768), a1);
        ao = WMMA_BF16(xa[kc], load_frag(bo + (size_t)kc * 32768), ao);
      }
      const float bb1 = b1p[nt * 16 + nC];
      const float bbo = b1o[nt * 16 + nC];
#pragma unroll
      for (int r = 0; r < 8; ++r) {
        const int m = r + 8 * halfk;  // C layout: lanes>=16 hold rows 8..15
        h1b[m * 40 + 16 * sub + nC] = f2bf(fmaxf(a1[r] + bb1, 0.0f));
        hxb[m * HXP + nt * 16 + nC] = f2bf(ao[r] + bbo);
      }
    }
    // restage h1 (16x32) into A-fragment and accumulate logits
    v16bf a2;
    {
      const unsigned short* hp = h1b + mA * 40;
      Frag32B pr;
      pr.a = *(const v4i*)(hp + o);
      pr.b = *(const v4i*)(hp + o + 16);
      a2 = __builtin_bit_cast(v16bf, pr);
    }
    acc2p = WMMA_BF16(a2, load_frag(fW2p + (size_t)c * 1024 + lane * 32), acc2p);
  }

  // ---- softmax over 8 classes (cross-lane within octets) ----
  float pi[8];
  {
    const float bb = (nC < 8) ? b2p[nC] : -3.0e38f;
#pragma unroll
    for (int r = 0; r < 8; ++r) {
      float z = acc2p[r] + bb;
      float mx = z;
      mx = fmaxf(mx, __shfl_xor(mx, 1, 32));
      mx = fmaxf(mx, __shfl_xor(mx, 2, 32));
      mx = fmaxf(mx, __shfl_xor(mx, 4, 32));
      float e = __expf(z - mx);
      float s = e;
      s += __shfl_xor(s, 1, 32);
      s += __shfl_xor(s, 2, 32);
      s += __shfl_xor(s, 4, 32);
      pi[r] = e / s;
    }
  }

  // ---- f-stage: 15 quadrature nodes in 3 groups of 5 ----
  const float tA = t[row0 + mA];
  const float bo2 = (nC < 8) ? b2o[nC] : 0.0f;
  float predacc[8];
#pragma unroll
  for (int r = 0; r < 8; ++r) predacc[r] = 0.0f;

  for (int g = 0; g < 3; ++g) {
    float tau[5];
    v8f accf[5];
#pragma unroll
    for (int n = 0; n < 5; ++n) {
      tau[n] = tA * 0.5f * (1.0f + UQ[g * 5 + n]);
      accf[n] = (v8f){0, 0, 0, 0, 0, 0, 0, 0};
    }
    for (int c = 0; c < 16; ++c) {
      float hxf[16], wlf[16];
      {
        const unsigned short* hp = hxb + mA * HXP + c * 32;
        v4i q0 = *(const v4i*)(hp + o);
        v4i q1 = *(const v4i*)(hp + o + 16);
#pragma unroll
        for (int j = 0; j < 4; ++j) {
          hxf[2 * j] = bf2f((unsigned)q0[j] & 0xFFFFu);
          hxf[2 * j + 1] = bf2f((unsigned)q0[j] >> 16);
          hxf[8 + 2 * j] = bf2f((unsigned)q1[j] & 0xFFFFu);
          hxf[8 + 2 * j + 1] = bf2f((unsigned)q1[j] >> 16);
        }
      }
      {
        const float* wp = wlast + c * 32 + o;
        v4f w0 = *(const v4f*)(wp);
        v4f w1 = *(const v4f*)(wp + 4);
        v4f w2 = *(const v4f*)(wp + 16);
        v4f w3 = *(const v4f*)(wp + 20);
#pragma unroll
        for (int j = 0; j < 4; ++j) {
          wlf[j] = w0[j];
          wlf[4 + j] = w1[j];
          wlf[8 + j] = w2[j];
          wlf[12 + j] = w3[j];
        }
      }
      const v16bf bfrag = load_frag(fW2o + (size_t)c * 1024 + lane * 32);
#pragma unroll
      for (int n = 0; n < 5; ++n) {
        v16u u;
#pragma unroll
        for (int i = 0; i < 16; ++i) {
          const float hv = fmaf(tau[n], wlf[i], hxf[i]);
          u[i] = f2bf(fmaxf(hv, 0.0f));
        }
        accf[n] = WMMA_BF16(__builtin_bit_cast(v16bf, u), bfrag, accf[n]);
      }
    }
#pragma unroll
    for (int n = 0; n < 5; ++n) {
      const float wq = WQ[g * 5 + n];
#pragma unroll
      for (int r = 0; r < 8; ++r) {
        const float v = accf[n][r] + bo2;
        const float sp = fmaxf(v, 0.0f) + __logf(1.0f + __expf(-fabsf(v)));
        predacc[r] = fmaf(wq, sp, predacc[r]);
      }
    }
  }

  // ---- CIF + outputs: preds then pi, each (B,8) row-major ----
#pragma unroll
  for (int r = 0; r < 8; ++r) {
    const int m = row0 + r + 8 * halfk;
    const float pr = t[m] * 0.5f * predacc[r];
    const float cif = 1.0f - __expf(-pr);
    if (nC < 8) {
      out[(size_t)m * 8 + nC] = pi[r] * cif;
      out[(size_t)16384 * 8 + (size_t)m * 8 + nC] = pi[r];
    }
  }
}

// ---------------------------------------------------------------------------
extern "C" void kernel_launch(void* const* d_in, const int* in_sizes, int n_in,
                              void* d_out, int out_size, void* d_ws, size_t ws_size,
                              hipStream_t stream) {
  (void)in_sizes; (void)n_in; (void)out_size; (void)ws_size;
  const float* x = (const float*)d_in[0];
  const float* t = (const float*)d_in[1];
  const float* W1p = (const float*)d_in[2];
  const float* b1p = (const float*)d_in[3];
  const float* W2p = (const float*)d_in[4];
  const float* b2p = (const float*)d_in[5];
  const float* W1o = (const float*)d_in[6];
  const float* b1o = (const float*)d_in[7];
  const float* W2o = (const float*)d_in[8];
  const float* b2o = (const float*)d_in[9];
  float* out = (float*)d_out;
  unsigned char* ws = (unsigned char*)d_ws;

  const int prep_elems = 2 * 131072 + 2 * 8192 + 512;  // 279040
  odesurv_prep<<<(prep_elems + 255) / 256, 256, 0, stream>>>(W1p, W2p, W1o, W2o, ws);
  // 16384 rows / 16 per tile = 1024 tiles; 2 waves per block
  odesurv_main<<<512, 64, 0, stream>>>(x, t, b1p, b2p, b1o, b2o, ws, out);
}
